// TwoFWLConv_33844342293141
// MI455X (gfx1250) — matrix operand order repaired and checked by reading, not compiled, not run on hardware
//
#include <hip/hip_runtime.h>

typedef __attribute__((ext_vector_type(16))) __bf16 v16bf;
typedef __attribute__((ext_vector_type(8)))  __bf16 v8bf;
typedef __attribute__((ext_vector_type(4)))  __bf16 v4bf;
typedef __attribute__((ext_vector_type(8)))  float  v8f;
typedef __attribute__((ext_vector_type(4)))  float  v4f;

#define NNZ   131072
#define EMB   128
#define MTRI  1048576
#define WSTRIDE 136   // padded LDS row stride (bf16 elems): 272B rows -> banks spread, 16B aligned

__device__ __forceinline__ v16bf cat8(v8bf lo, v8bf hi) {
  return __builtin_shufflevector(lo, hi, 0,1,2,3,4,5,6,7,8,9,10,11,12,13,14,15);
}

// Transpose-load a 128x128 f32 weight into LDS as bf16 [col][row], padded stride.
__device__ __forceinline__ void load_w_T(const float* __restrict__ W, __bf16* Ws, int tid) {
  for (int i = tid; i < EMB * EMB; i += 128) {
    int r = i >> 7, c = i & 127;
    Ws[c * WSTRIDE + r] = (__bf16)W[i];
  }
}

// A fragment (16x32 bf16) straight from global f32 rows, as 4x b128 loads.
// lane<16: a[0..7]=K[kb..kb+7], a[8..15]=K[kb+16..kb+23]; lane>=16: +8 / +24.
__device__ __forceinline__ v16bf a_frag_global(const float* __restrict__ X,
                                               int grow, int kb, int laneHi) {
  const float* p = X + (size_t)grow * EMB + kb + (laneHi << 3);  // 16B aligned
  v4f f0 = *(const v4f*)(p);
  v4f f1 = *(const v4f*)(p + 4);
  v4f f2 = *(const v4f*)(p + 16);
  v4f f3 = *(const v4f*)(p + 20);
  v16bf a;
#pragma unroll
  for (int j = 0; j < 4; ++j) {
    a[j]      = (__bf16)f0[j];
    a[j + 4]  = (__bf16)f1[j];
    a[j + 8]  = (__bf16)f2[j];
    a[j + 12] = (__bf16)f3[j];
  }
  return a;
}

// A fragment from LDS (hidden activations, bf16, padded rows).
__device__ __forceinline__ v16bf a_frag_lds(const __bf16* Hrow, int kb, int laneHi) {
  const __bf16* p = Hrow + kb + (laneHi << 3);
  v8bf lo = *(const v8bf*)p;          // 16B aligned
  v8bf hi = *(const v8bf*)(p + 16);   // 16B aligned
  return cat8(lo, hi);
}

// B fragment (32x16 bf16) from transposed LDS weight.
// lane<16: b[0..7]=W[kb..kb+7][col], b[8..15]=W[kb+16..kb+23][col]; lane>=16: +8 / +24.
__device__ __forceinline__ v16bf b_frag(const __bf16* Ws, int col, int kb, int laneHi) {
  const __bf16* p = Ws + col * WSTRIDE + kb + (laneHi << 3);
  v8bf lo = *(const v8bf*)p;
  v8bf hi = *(const v8bf*)(p + 16);
  return cat8(lo, hi);
}

// One fused GEMM stage: 16 rows (per wave) x 128 cols, K=128 in 4 WMMA steps.
// Epilogue: bias + ReLU, result to LDS (bf16, padded) or global (bf16, dense).
template <bool A_FROM_LDS, bool OUT_TO_LDS>
__device__ __forceinline__ void gemm_tile(const float* Xg,
                                          const __bf16* Hsrc, __bf16* Hdst,
                                          const __bf16* Ws,
                                          const float* __restrict__ bias,
                                          __bf16* Yg,
                                          int rowBase, int wave, int lane) {
  const int lmod = lane & 15, laneHi = lane >> 4;
  const int localRow = wave * 16 + lmod;

  v16bf afr[4];
#pragma unroll
  for (int k = 0; k < 4; ++k) {
    if (A_FROM_LDS) afr[k] = a_frag_lds(Hsrc + localRow * WSTRIDE, k * 32, laneHi);
    else            afr[k] = a_frag_global(Xg, rowBase + localRow, k * 32, laneHi);
  }

#pragma unroll
  for (int nt = 0; nt < 8; ++nt) {
    v8f c = {};
#pragma unroll
    for (int k = 0; k < 4; ++k) {
      v16bf b = b_frag(Ws, nt * 16 + lmod, k * 32, laneHi);
      c = __builtin_amdgcn_wmma_f32_16x16x32_bf16(false, afr[k], false, b,
                                                  (short)0, c, false, false);
    }
    const int col = nt * 16 + lmod;
    const float bv = bias[col];
    const int r0 = wave * 16 + (laneHi << 3);   // C layout: VGPR v -> row r0+v
#pragma unroll
    for (int v = 0; v < 8; ++v) {
      float y = c[v] + bv;
      y = fmaxf(y, 0.0f);
      if (OUT_TO_LDS) Hdst[(r0 + v) * WSTRIDE + col] = (__bf16)y;
      else Yg[(size_t)(rowBase + r0 + v) * EMB + col] = (__bf16)y;
    }
  }
}

__global__ void __launch_bounds__(128)
mlp2_wmma_kernel(const float* __restrict__ Xv,
                 const float* __restrict__ W10, const float* __restrict__ b10,
                 const float* __restrict__ W11, const float* __restrict__ b11,
                 const float* __restrict__ W20, const float* __restrict__ b20,
                 const float* __restrict__ W21, const float* __restrict__ b21,
                 __bf16* __restrict__ X1, __bf16* __restrict__ X2) {
  extern __shared__ __bf16 smem[];
  __bf16* Ws = smem;                  // 128 x WSTRIDE : transposed weight
  __bf16* Hs = smem + 128 * WSTRIDE;  // 64  x WSTRIDE : hidden activations

  const int tid  = threadIdx.x;
  const int wave = tid >> 5;
  const int lane = tid & 31;
  const int rowBase = blockIdx.x * 64;

  // --- MLP1: h = relu(X W10 + b10); X1 = relu(h W11 + b11) ---
  load_w_T(W10, Ws, tid);
  __syncthreads();
  gemm_tile<false, true >(Xv, Hs, Hs, Ws, b10, nullptr, rowBase, wave, lane);
  __syncthreads();
  load_w_T(W11, Ws, tid);
  __syncthreads();
  gemm_tile<true,  false>(nullptr, Hs, nullptr, Ws, b11, X1, rowBase, wave, lane);
  __syncthreads();

  // --- MLP2: h = relu(X W20 + b20); X2 = relu(h W21 + b21) ---
  load_w_T(W20, Ws, tid);
  __syncthreads();
  gemm_tile<false, true >(Xv, Hs, Hs, Ws, b20, nullptr, rowBase, wave, lane);
  __syncthreads();
  load_w_T(W21, Ws, tid);
  __syncthreads();
  gemm_tile<true,  false>(nullptr, Hs, nullptr, Ws, b21, X2, rowBase, wave, lane);
}

// One wave per triple: gather two bf16 rows (4 elems/lane), f32 multiply,
// f32 atomic scatter-add into out[ind_out].
__global__ void __launch_bounds__(256)
scatter_kernel(const __bf16* __restrict__ X1, const __bf16* __restrict__ X2,
               const int* __restrict__ ia, const int* __restrict__ ib,
               const int* __restrict__ io, float* __restrict__ out) {
  const int lane = threadIdx.x & 31;
  const int t = blockIdx.x * 8 + (threadIdx.x >> 5);
  if (t >= MTRI) return;
  const int a = ia[t], b = ib[t], o = io[t];
  const int e = lane * 4;
  v4bf x1 = *(const v4bf*)(X1 + (size_t)a * EMB + e);
  v4bf x2 = *(const v4bf*)(X2 + (size_t)b * EMB + e);
  float* op = out + (size_t)o * EMB + e;
#pragma unroll
  for (int j = 0; j < 4; ++j) {
    atomicAdd(op + j, (float)x1[j] * (float)x2[j]);
  }
}

__global__ void __launch_bounds__(256)
zero_kernel(float* __restrict__ out, int n4) {
  int i = blockIdx.x * 256 + threadIdx.x;
  if (i < n4) {
    v4f z = {0.f, 0.f, 0.f, 0.f};
    ((v4f*)out)[i] = z;
  }
}

extern "C" void kernel_launch(void* const* d_in, const int* in_sizes, int n_in,
                              void* d_out, int out_size, void* d_ws, size_t ws_size,
                              hipStream_t stream) {
  const float* Xv  = (const float*)d_in[0];
  const float* W10 = (const float*)d_in[1];
  const float* b10 = (const float*)d_in[2];
  const float* W11 = (const float*)d_in[3];
  const float* b11 = (const float*)d_in[4];
  const float* W20 = (const float*)d_in[5];
  const float* b20 = (const float*)d_in[6];
  const float* W21 = (const float*)d_in[7];
  const float* b21 = (const float*)d_in[8];
  const int* ia = (const int*)d_in[9];
  const int* ib = (const int*)d_in[10];
  const int* io = (const int*)d_in[11];
  float* out = (float*)d_out;

  __bf16* X1 = (__bf16*)d_ws;               // 32 MB
  __bf16* X2 = X1 + (size_t)NNZ * EMB;      // 32 MB

  const int n4 = out_size / 4;
  zero_kernel<<<(n4 + 255) / 256, 256, 0, stream>>>(out, n4);

  const size_t smem = (size_t)(128 * WSTRIDE + 64 * WSTRIDE) * sizeof(__bf16); // 52224 B
  mlp2_wmma_kernel<<<NNZ / 64, 128, smem, stream>>>(Xv, W10, b10, W11, b11,
                                                    W20, b20, W21, b21, X1, X2);

  scatter_kernel<<<MTRI / 8, 256, 0, stream>>>(X1, X2, ia, ib, io, out);
}
